// NonLocal_81698867904587
// MI455X (gfx1250) — matrix operand order repaired and checked by reading, not compiled, not run on hardware
//
#include <hip/hip_runtime.h>
#include <hip/hip_bf16.h>

#define LOG2E 1.44269504088896340736f

typedef __attribute__((ext_vector_type(16))) _Float16 v16h;
typedef __attribute__((ext_vector_type(8)))  float    v8f;

static constexpr int B_ = 4, C_ = 256, H_ = 96, W_ = 96;
static constexpr int N_ = H_ * W_;       // 9216
static constexpr int D_ = 128;
static constexpr int BN_ = B_ * N_;      // 36864

union F16Frag { v16h h; uint4 u[2]; };

// A-fragment style: two 16B chunks (elements 0..7 from p0, 8..15 from p1)
__device__ __forceinline__ v16h frag_2x8(const _Float16* p0, const _Float16* p1) {
  F16Frag f;
  f.u[0] = *(const uint4*)p0;
  f.u[1] = *(const uint4*)p1;
  return f.h;
}
// B-fragment style: 16 contiguous halves
__device__ __forceinline__ v16h frag_16(const _Float16* p) {
  F16Frag f;
  f.u[0] = *(const uint4*)p;
  f.u[1] = *(const uint4*)(p + 8);
  return f.h;
}

__device__ __forceinline__ v8f wmma_f16(v16h a, v16h b, v8f c) {
  return __builtin_amdgcn_wmma_f32_16x16x32_f16(false, a, false, b, (short)0, c,
                                                false, false);
}

// XOR-swizzle reductions within 16-lane halves (group-of-32 mode:
// offset = and_mask(0x1f) | xor_mask<<10)
__device__ __forceinline__ float red16_max(float v) {
  v = fmaxf(v, __int_as_float(__builtin_amdgcn_ds_swizzle(__float_as_int(v), 0x041f)));
  v = fmaxf(v, __int_as_float(__builtin_amdgcn_ds_swizzle(__float_as_int(v), 0x081f)));
  v = fmaxf(v, __int_as_float(__builtin_amdgcn_ds_swizzle(__float_as_int(v), 0x101f)));
  v = fmaxf(v, __int_as_float(__builtin_amdgcn_ds_swizzle(__float_as_int(v), 0x201f)));
  return v;
}
__device__ __forceinline__ float red16_sum(float v) {
  v += __int_as_float(__builtin_amdgcn_ds_swizzle(__float_as_int(v), 0x041f));
  v += __int_as_float(__builtin_amdgcn_ds_swizzle(__float_as_int(v), 0x081f));
  v += __int_as_float(__builtin_amdgcn_ds_swizzle(__float_as_int(v), 0x101f));
  v += __int_as_float(__builtin_amdgcn_ds_swizzle(__float_as_int(v), 0x201f));
  return v;
}

// ---- CDNA5 async global->LDS copy (ASYNCcnt path, no VGPR data movement) ----
// Inline asm per CDNA5 ISA: VDST = per-lane LDS byte
// address, VADDR = 64-bit global address (GV mode).
__device__ __forceinline__ void async_copy16(_Float16* lds_dst, const _Float16* gsrc) {
  unsigned lo =
      (unsigned)(uintptr_t)(__attribute__((address_space(3))) void*)lds_dst;
  asm volatile("global_load_async_to_lds_b128 %0, %1, off"
               :: "v"(lo), "v"((unsigned long long)(uintptr_t)gsrc)
               : "memory");
}
__device__ __forceinline__ void wait_async0() {
  asm volatile("s_wait_asynccnt 0x0" ::: "memory");
}

// ---------------------------------------------------------------------------
// Kernel 0: convert weights fp32 -> f16 once. theta weights folded with log2e
// so attention logits are already in the exp2 domain.
// ---------------------------------------------------------------------------
__global__ void k_cvt_w(const float* __restrict__ wt, const float* __restrict__ wp,
                        const float* __restrict__ wg, const float* __restrict__ wo,
                        _Float16* __restrict__ w16, _Float16* __restrict__ wo16) {
  const int DC = D_ * C_;
  int i = blockIdx.x * blockDim.x + threadIdx.x;
  if (i < DC) {
    w16[i]          = (_Float16)(wt[i] * LOG2E);
    w16[DC + i]     = (_Float16)wp[i];
    w16[2 * DC + i] = (_Float16)wg[i];
    wo16[i]         = (_Float16)wo[i];
  }
}

// ---------------------------------------------------------------------------
// Kernel 1: projections. Block = 128 threads = 4 waves, each wave does a
// 16-row x 384-col GEMM tile with K=256. x tile transposed into LDS as f16.
// Outputs: theta[BN][D], phi[BN][D] row-major; g stored transposed gT[B][D][N].
// ---------------------------------------------------------------------------
__global__ __launch_bounds__(128) void k_proj(
    const float* __restrict__ x, const _Float16* __restrict__ w16,
    const float* __restrict__ bt, const float* __restrict__ bp,
    const float* __restrict__ bg,
    _Float16* __restrict__ theta, _Float16* __restrict__ phi,
    _Float16* __restrict__ gT) {
  __shared__ _Float16 lds_x[64 * 264];  // [n_local][c], stride 264 (16B aligned)
  const int tid  = threadIdx.x;
  const int wave = tid >> 5;
  const int lane = tid & 31;
  const int ln   = lane & 15;
  const int hi   = lane >> 4;
  const long rowBase = (long)blockIdx.x * 64;  // flat row in [0, BN)
  const int  b  = (int)(rowBase / N_);
  const int  n0 = (int)(rowBase % N_);

  {  // cooperative transpose-load of x tile
    const float* xb = x + (size_t)b * C_ * N_;
    for (int i = tid; i < 64 * C_; i += 128) {
      int nl = i & 63;
      int c  = i >> 6;
      lds_x[nl * 264 + c] = (_Float16)xb[(size_t)c * N_ + n0 + nl];
    }
  }
  __syncthreads();

  // A fragments for this wave's 16 rows, all of K=256 (8 K-steps)
  v16h A[8];
  const int rloc = wave * 16 + ln;
#pragma unroll
  for (int kk = 0; kk < 8; ++kk) {
    const _Float16* p = &lds_x[rloc * 264 + kk * 32 + (hi ? 16 : 0)];
    A[kk] = frag_16(p);
  }

  for (int ct = 0; ct < 24; ++ct) {          // 24 x 16 = 384 output cols
    const int sel = ct >> 3;                 // 0=theta 1=phi 2=g
    const int d   = ((ct & 7) << 4) + ln;    // col within D
    const _Float16* wsel = w16 + (size_t)sel * D_ * C_ + (size_t)d * C_ + (hi ? 16 : 0);
    v8f acc = {};
#pragma unroll
    for (int kk = 0; kk < 8; ++kk)
      acc = wmma_f16(A[kk], frag_16(wsel + kk * 32), acc);

    const float* bptr = (sel == 0) ? bt : ((sel == 1) ? bp : bg);
    const float bias  = bptr[d] * ((sel == 0) ? LOG2E : 1.0f);
#pragma unroll
    for (int i = 0; i < 8; ++i) {
      long gr = rowBase + wave * 16 + i + hi * 8;
      _Float16 hv = (_Float16)(acc[i] + bias);
      if (sel == 0)      theta[gr * D_ + d] = hv;
      else if (sel == 1) phi[gr * D_ + d]   = hv;
      else {
        int n = (int)(gr % N_);
        gT[((size_t)b * D_ + d) * N_ + n] = hv;
      }
    }
  }
}

// ---------------------------------------------------------------------------
// Kernel 2: flash attention. Each wave owns 16 queries, streams all 9216 keys
// in chunks of 32 with online softmax. phi/gT chunks are double-buffered in
// LDS and fetched with GLOBAL_LOAD_ASYNC_TO_LDS_B128 (ASYNCcnt), so the next
// chunk streams in behind the current chunk's WMMAs.
// ---------------------------------------------------------------------------
__global__ __launch_bounds__(128) void k_attn(
    const _Float16* __restrict__ theta, const _Float16* __restrict__ phi,
    const _Float16* __restrict__ gT, _Float16* __restrict__ y) {
  __shared__ _Float16 lds_phi[2][32 * 136];  // [buf][key_local][c], stride 136
  __shared__ _Float16 lds_g[2][128 * 40];    // [buf][d][key_local], stride 40
  __shared__ _Float16 lds_p[4 * 16 * 40];    // per-wave P transpose scratch

  const int tid  = threadIdx.x;
  const int wave = tid >> 5;
  const int lane = tid & 31;
  const int ln   = lane & 15;
  const int hi   = lane >> 4;
  const long rowBase = (long)blockIdx.x * 64 + wave * 16;  // flat query base
  const int  b = (int)(rowBase / N_);
  const size_t phiBase = (size_t)b * N_ * D_;
  const size_t gBase   = (size_t)b * D_ * N_;
  _Float16* ldsp = lds_p + wave * 16 * 40;

  // per-thread async-copy coordinates (8 x 16B per thread per chunk)
  const int pr = tid >> 4, ps = tid & 15;  // phi: rows pr+8j, seg ps
  const int gr = tid >> 2, gs = tid & 3;   // gT : rows gr+32j, seg gs

  auto issue_chunk = [&](int kb, int buf) {
#pragma unroll
    for (int j = 0; j < 4; ++j) {
      const _Float16* src = phi + phiBase + (size_t)(kb + pr + 8 * j) * D_ + ps * 8;
      async_copy16(&lds_phi[buf][(pr + 8 * j) * 136 + ps * 8], src);
    }
#pragma unroll
    for (int j = 0; j < 4; ++j) {
      const _Float16* src = gT + gBase + (size_t)(gr + 32 * j) * N_ + kb + gs * 8;
      async_copy16(&lds_g[buf][(gr + 32 * j) * 40 + gs * 8], src);
    }
  };

  // theta A fragments (loop-invariant, K=128 in 4 steps of 32)
  v16h At[4];
#pragma unroll
  for (int kk = 0; kk < 4; ++kk) {
    const _Float16* p = theta + (rowBase + ln) * D_ + kk * 32 + (hi ? 16 : 0);
    At[kk] = frag_16(p);
  }

  v8f O[8] = {};
  float m[8], l[8];
#pragma unroll
  for (int i = 0; i < 8; ++i) { m[i] = -1e30f; l[i] = 0.0f; }

  issue_chunk(0, 0);
  for (int kb = 0, it = 0; kb < N_; kb += 32, ++it) {
    const int buf = it & 1;
    wait_async0();    // this wave's loads for `buf` have landed in LDS
    __syncthreads();  // everyone's landed; everyone done reading buf^1
    if (kb + 32 < N_) issue_chunk(kb + 32, buf ^ 1);  // prefetch behind compute

    // S = theta @ phi^T : two 16x16 tiles (32 keys), K=128
    const _Float16* lphi = lds_phi[buf];
    const _Float16* lg   = lds_g[buf];
    v8f s0 = {}, s1 = {};
#pragma unroll
    for (int kk = 0; kk < 4; ++kk) {
      s0 = wmma_f16(At[kk], frag_16(&lphi[ln * 136 + kk * 32 + hi * 16]), s0);
      s1 = wmma_f16(At[kk], frag_16(&lphi[(16 + ln) * 136 + kk * 32 + hi * 16]), s1);
    }

    // online softmax update (logits already in log2 domain)
    float corr[8];
#pragma unroll
    for (int i = 0; i < 8; ++i) {
      float rm = red16_max(fmaxf(s0[i], s1[i]));
      float mn = fmaxf(m[i], rm);
      float cc = __builtin_amdgcn_exp2f(m[i] - mn);
      m[i] = mn;
      float p0 = __builtin_amdgcn_exp2f(s0[i] - mn);
      float p1 = __builtin_amdgcn_exp2f(s1[i] - mn);
      s0[i] = p0; s1[i] = p1;
      l[i] = l[i] * cc + red16_sum(p0 + p1);
      corr[i] = cc;
    }
#pragma unroll
    for (int t = 0; t < 8; ++t)
#pragma unroll
      for (int i = 0; i < 8; ++i) O[t][i] *= corr[i];

    // P: D-layout -> A-layout via per-wave LDS scratch (in-wave LDS is in-order)
#pragma unroll
    for (int i = 0; i < 8; ++i) {
      int r = i + hi * 8;
      ldsp[r * 40 + ln]      = (_Float16)s0[i];
      ldsp[r * 40 + 16 + ln] = (_Float16)s1[i];
    }
    v16h Pf;
    {
      const _Float16* p = &ldsp[ln * 40 + (hi ? 16 : 0)];
      Pf = frag_16(p);
    }
    // O += P @ g  (K=32, 8 output tiles across D)
#pragma unroll
    for (int t = 0; t < 8; ++t)
      O[t] = wmma_f16(Pf, frag_16(&lg[(t * 16 + ln) * 40 + hi * 16]), O[t]);
  }

  float inv[8];
#pragma unroll
  for (int i = 0; i < 8; ++i) inv[i] = 1.0f / l[i];
#pragma unroll
  for (int t = 0; t < 8; ++t) {
#pragma unroll
    for (int i = 0; i < 8; ++i) {
      long grr = rowBase + i + hi * 8;
      y[grr * D_ + t * 16 + ln] = (_Float16)(O[t][i] * inv[i]);
    }
  }
}

// ---------------------------------------------------------------------------
// Kernel 3: z = y @ w_out^T + b_out + x, written NCHW with an LDS transpose
// so stores are 64B-contiguous along n.
// ---------------------------------------------------------------------------
__global__ __launch_bounds__(128) void k_out(
    const _Float16* __restrict__ y, const _Float16* __restrict__ wo16,
    const float* __restrict__ bo, const float* __restrict__ x,
    float* __restrict__ z) {
  __shared__ float lds_t[4 * 16 * 17];
  const int tid  = threadIdx.x;
  const int wave = tid >> 5;
  const int lane = tid & 31;
  const int ln   = lane & 15;
  const int hi   = lane >> 4;
  const long rowBase = (long)blockIdx.x * 64 + wave * 16;
  const int  b  = (int)(rowBase / N_);
  const int  n0 = (int)(rowBase % N_);
  float* ldt = lds_t + wave * 16 * 17;

  v16h A[4];
#pragma unroll
  for (int kk = 0; kk < 4; ++kk) {
    const _Float16* p = y + (rowBase + ln) * D_ + kk * 32 + (hi ? 16 : 0);
    A[kk] = frag_16(p);
  }

  for (int ct = 0; ct < 16; ++ct) {       // 16 x 16 = 256 output channels
    const int c = ct * 16 + ln;
    const _Float16* wp = wo16 + (size_t)c * D_ + (hi ? 16 : 0);
    v8f acc = {};
#pragma unroll
    for (int kk = 0; kk < 4; ++kk)
      acc = wmma_f16(A[kk], frag_16(wp + kk * 32), acc);
    const float bias = bo[c];
    // transpose: ldt[col][row]
#pragma unroll
    for (int i = 0; i < 8; ++i)
      ldt[ln * 17 + i + hi * 8] = acc[i] + bias;
    __syncthreads();
    {  // coalesced residual-add + store: 2 lanes per channel row
      int j = lane >> 1, h = lane & 1;
      size_t g = ((size_t)b * C_ + ct * 16 + j) * N_ + n0 + h * 8;
      const float* lr = &ldt[j * 17 + h * 8];
#pragma unroll
      for (int e = 0; e < 8; ++e) z[g + e] = lr[e] + x[g + e];
    }
    __syncthreads();
  }
}

// ---------------------------------------------------------------------------
extern "C" void kernel_launch(void* const* d_in, const int* in_sizes, int n_in,
                              void* d_out, int out_size, void* d_ws, size_t ws_size,
                              hipStream_t stream) {
  const float* x  = (const float*)d_in[0];
  const float* wt = (const float*)d_in[1];
  const float* bt = (const float*)d_in[2];
  const float* wp = (const float*)d_in[3];
  const float* bp = (const float*)d_in[4];
  const float* wg = (const float*)d_in[5];
  const float* bg = (const float*)d_in[6];
  const float* wo = (const float*)d_in[7];
  const float* bo = (const float*)d_in[8];
  float* z = (float*)d_out;

  _Float16* ws = (_Float16*)d_ws;
  const size_t BND = (size_t)BN_ * D_;
  _Float16* theta = ws;                    // [BN][D]
  _Float16* phi   = theta + BND;           // [BN][D]
  _Float16* gT    = phi + BND;             // [B][D][N]
  _Float16* yb    = gT + BND;              // [BN][D]
  _Float16* w16   = yb + BND;              // [3][D][C]
  _Float16* wo16  = w16 + 3 * (size_t)D_ * C_;  // [C][D]

  k_cvt_w<<<dim3((D_ * C_ + 255) / 256), dim3(256), 0, stream>>>(wt, wp, wg, wo, w16, wo16);
  k_proj<<<dim3(BN_ / 64), dim3(128), 0, stream>>>(x, w16, bt, bp, bg, theta, phi, gT);
  k_attn<<<dim3(BN_ / 64), dim3(128), 0, stream>>>(theta, phi, gT, yb);
  k_out<<<dim3(BN_ / 64), dim3(128), 0, stream>>>(yb, wo16, bo, x, z);
}
